// AttentiveFP_Simple_32555852103853
// MI455X (gfx1250) — matrix operand order repaired and checked by reading, not compiled, not run on hardware
//
#include <hip/hip_runtime.h>
#include <hip/hip_bf16.h>

typedef __attribute__((ext_vector_type(16))) _Float16 v16h;
typedef __attribute__((ext_vector_type(8)))  float    v8f;

#define THREADS 256

__device__ inline float atomicMaxF(float* addr, float v) {
  // sign-aware int/uint trick: valid for non-NaN floats
  if (v >= 0.f)
    return __int_as_float(atomicMax((int*)addr, __float_as_int(v)));
  else
    return __uint_as_float(atomicMin((unsigned int*)addr, __float_as_uint(v)));
}

// ---------------------------------------------------------------------------
// Generic fill
__global__ __launch_bounds__(THREADS) void k_fill(float* __restrict__ p, float v, size_t n) {
  size_t i = (size_t)blockIdx.x * THREADS + threadIdx.x;
  if (i < n) p[i] = v;
}

// f32 -> f16 convert
__global__ __launch_bounds__(THREADS) void k_cvt_f16(const float* __restrict__ src,
                                                     _Float16* __restrict__ dst, size_t n) {
  size_t i = (size_t)blockIdx.x * THREADS + threadIdx.x;
  if (i < n) dst[i] = (_Float16)src[i];
}

// W [K,Nc] f32 row-major -> Wt [Npad,K] f16 (transposed, zero-padded cols)
__global__ __launch_bounds__(THREADS) void k_transpose_w(const float* __restrict__ W,
                                                         _Float16* __restrict__ Wt,
                                                         int K, int Nc, int Npad) {
  int i = blockIdx.x * THREADS + threadIdx.x;
  if (i >= Npad * K) return;
  int n = i / K, k = i - n * K;
  float v = (n < Nc) ? W[(size_t)k * Nc + n] : 0.f;
  Wt[i] = (_Float16)v;
}

// ---------------------------------------------------------------------------
// WMMA GEMM: Out[M,ldo] = A[M,K](f16) @ Bt[Npad,K]^T(f16) (+bias), f32 accum.
// Each wave computes a 16x64 output strip (4 N-tiles); A fragment reused
// across 4 v_wmma_f32_16x16x32_f16 per K-step. Branch-free body: out-of-range
// rows / tiles are CLAMPED for loads (safe, inside padded buffers) and the
// garbage results are simply not stored.
__global__ __launch_bounds__(THREADS) void k_wmma_gemm(
    const _Float16* __restrict__ A, const _Float16* __restrict__ Bt,
    const float* __restrict__ bias, float* __restrict__ Out,
    int M, int K, int Npad, int Nreal, int ldo) {
  const int lane = threadIdx.x & 31;
  const int wave = threadIdx.x >> 5;
  const int Mt  = (M + 15) >> 4;
  const int Nt  = Npad >> 4;
  const int Ntg = (Nt + 3) >> 2;           // groups of 4 N-tiles
  int t = blockIdx.x * 8 + wave;
  if (t >= Mt * Ntg) return;
  const int tm  = t / Ntg;
  const int tn0 = (t - tm * Ntg) * 4;

  const int l15 = lane & 15;
  const int hi  = (lane & 16) ? 1 : 0;
  const int akb = hi ? 8 : 0;              // A k-base per half-wave
  const int bkb = hi ? 8 : 0;              // B k-base per half-wave

  int arow = tm * 16 + l15;                // A: lane holds one row
  if (arow > M - 1) arow = M - 1;          // clamp: garbage rows never stored
  const _Float16* Ar = A + (size_t)arow * K;

  const _Float16* Br[4];
#pragma unroll
  for (int j = 0; j < 4; ++j) {
    int jt = tn0 + j;
    if (jt > Nt - 1) jt = Nt - 1;          // clamp: garbage tiles never stored
    Br[j] = Bt + (size_t)(jt * 16 + l15) * K;
  }

  v8f acc[4];
#pragma unroll
  for (int j = 0; j < 4; ++j)
    acc[j] = (v8f){0.f, 0.f, 0.f, 0.f, 0.f, 0.f, 0.f, 0.f};

  for (int k0 = 0; k0 < K; k0 += 32) {
    union { v16h v; uint4 q[2]; } a;
    a.q[0] = *(const uint4*)(Ar + k0 + akb);
    a.q[1] = *(const uint4*)(Ar + k0 + akb + 16);
#pragma unroll
    for (int j = 0; j < 4; ++j) {
      union { v16h v; uint4 q[2]; } b;
      b.q[0] = *(const uint4*)(Br[j] + k0 + bkb);
      b.q[1] = *(const uint4*)(Br[j] + k0 + bkb + 16);
      acc[j] = __builtin_amdgcn_wmma_f32_16x16x32_f16(
          false, a.v, false, b.v, (short)0, acc[j], false, false);
    }
  }

  // ---- store (guards only here)
  const int mbase = tm * 16 + (hi ? 8 : 0);
#pragma unroll
  for (int j = 0; j < 4; ++j) {
    if (tn0 + j < Nt) {
      const int ncol = (tn0 + j) * 16 + l15;
      if (ncol < Nreal) {
        float badd = (bias != nullptr) ? bias[ncol] : 0.f;
        for (int v = 0; v < 8; ++v) {
          int m = mbase + v;
          if (m < M) Out[(size_t)m * ldo + ncol] = acc[j][v] + badd;
        }
      }
    }
  }
}

// ---------------------------------------------------------------------------
// Per-(node,head) attention dot products: es/ed [N,4]
__global__ __launch_bounds__(THREADS) void k_head_dots(
    const float* __restrict__ xw, const float* __restrict__ a_s,
    const float* __restrict__ a_d, float* __restrict__ es, float* __restrict__ ed, int n) {
  int i = blockIdx.x * THREADS + threadIdx.x;
  if (i >= n * 4) return;
  int node = i >> 2, h = i & 3;
  const float* p = xw + (size_t)node * 256 + h * 64;
  const float* s = a_s + h * 64;
  const float* d = a_d + h * 64;
  float accs = 0.f, accd = 0.f;
#pragma unroll 8
  for (int c = 0; c < 64; ++c) { float v = p[c]; accs += v * s[c]; accd += v * d[c]; }
  es[i] = accs; ed[i] = accd;
}

__device__ inline void edge_pair(const int* __restrict__ ei, int E, int i, int& s, int& d) {
  if (i < E) { s = ei[i]; d = ei[E + i]; } else { s = d = i - E; } // self-loops
}

// segment max of leakyrelu(e) over dst
__global__ __launch_bounds__(THREADS) void k_edge_max(
    const int* __restrict__ ei, const float* __restrict__ es, const float* __restrict__ ed,
    float* __restrict__ mbuf, int E, int Etot) {
  int idx = blockIdx.x * THREADS + threadIdx.x;
  if (idx >= Etot * 4) return;
  int i = idx >> 2, h = idx & 3;
  int s, d; edge_pair(ei, E, i, s, d);
  float e = es[s * 4 + h] + ed[d * 4 + h];
  e = (e > 0.f) ? e : 0.2f * e;
  atomicMaxF(&mbuf[d * 4 + h], e);
}

// ex = exp(e - max); den += ex
__global__ __launch_bounds__(THREADS) void k_edge_exp(
    const int* __restrict__ ei, const float* __restrict__ es, const float* __restrict__ ed,
    const float* __restrict__ mbuf, float* __restrict__ ex, float* __restrict__ den,
    int E, int Etot) {
  int idx = blockIdx.x * THREADS + threadIdx.x;
  if (idx >= Etot * 4) return;
  int i = idx >> 2, h = idx & 3;
  int s, d; edge_pair(ei, E, i, s, d);
  float e = es[s * 4 + h] + ed[d * 4 + h];
  e = (e > 0.f) ? e : 0.2f * e;
  float v = __expf(e - mbuf[d * 4 + h]);
  ex[idx] = v;
  atomicAdd(&den[d * 4 + h], v);
}

// agg[dst] += alpha * xw[src]; block per edge, thread per channel
__global__ __launch_bounds__(THREADS) void k_edge_agg(
    const int* __restrict__ ei, const float* __restrict__ ex, const float* __restrict__ den,
    const float* __restrict__ xw, float* __restrict__ agg, int E, int Etot) {
  int i = blockIdx.x;
  if (i >= Etot) return;
  int c = threadIdx.x;           // 0..255
  int h = c >> 6;
  int s, d; edge_pair(ei, E, i, s, d);
  float alpha = ex[i * 4 + h] / den[d * 4 + h];
  atomicAdd(&agg[(size_t)d * 256 + c], alpha * xw[(size_t)s * 256 + c]);
}

// ---------------------------------------------------------------------------
// Column mean / rsqrt(var+eps) over rows (biased variance)
__global__ __launch_bounds__(THREADS) void k_col_stats(
    const float* __restrict__ X, int rows, int cols, float* __restrict__ mu,
    float* __restrict__ rs) {
  __shared__ float ssum[THREADS];
  __shared__ float ssq[THREADS];
  int col = blockIdx.x;
  float a = 0.f, b = 0.f;
  for (int r = threadIdx.x; r < rows; r += THREADS) {
    float v = X[(size_t)r * cols + col];
    a += v; b += v * v;
  }
  ssum[threadIdx.x] = a; ssq[threadIdx.x] = b;
  __syncthreads();
  for (int s = THREADS / 2; s > 0; s >>= 1) {
    if (threadIdx.x < s) { ssum[threadIdx.x] += ssum[threadIdx.x + s]; ssq[threadIdx.x] += ssq[threadIdx.x + s]; }
    __syncthreads();
  }
  if (threadIdx.x == 0) {
    float m = ssum[0] / (float)rows;
    float v = ssq[0] / (float)rows - m * m;
    mu[col] = m;
    rs[col] = rsqrtf(fmaxf(v, 0.f) + 1e-5f);
  }
}

// h = elu(bn(agg)) + res ; also emit f16 copy. cols is power of 2.
__global__ __launch_bounds__(THREADS) void k_gat_post(
    const float* __restrict__ agg, const float* __restrict__ res,
    const float* __restrict__ mu, const float* __restrict__ rs,
    const float* __restrict__ g, const float* __restrict__ be,
    float* __restrict__ h, _Float16* __restrict__ hf, size_t n, int colmask) {
  size_t i = (size_t)blockIdx.x * THREADS + threadIdx.x;
  if (i >= n) return;
  int c = (int)(i & (size_t)colmask);
  float z = (agg[i] - mu[c]) * rs[c] * g[c] + be[c];
  z = (z > 0.f) ? z : (__expf(z) - 1.f);
  float out = z + res[i];
  h[i] = out;
  hf[i] = (_Float16)out;
}

// relu(bn(x)) -> f16
__global__ __launch_bounds__(THREADS) void k_relu_post(
    const float* __restrict__ X, const float* __restrict__ mu, const float* __restrict__ rs,
    const float* __restrict__ g, const float* __restrict__ be,
    _Float16* __restrict__ outf, size_t n, int colmask) {
  size_t i = (size_t)blockIdx.x * THREADS + threadIdx.x;
  if (i >= n) return;
  int c = (int)(i & (size_t)colmask);
  float z = (X[i] - mu[c]) * rs[c] * g[c] + be[c];
  outf[i] = (_Float16)fmaxf(z, 0.f);
}

// ---------------------------------------------------------------------------
// graph pooling
__global__ __launch_bounds__(THREADS) void k_pool_acc(
    const float* __restrict__ h, const int* __restrict__ batch,
    float* __restrict__ sum, float* __restrict__ pmax, float* __restrict__ cnt, int N) {
  int i = blockIdx.x * THREADS + threadIdx.x;
  if (i >= N * 256) return;
  int node = i >> 8, c = i & 255;
  int g = batch[node];
  float v = h[i];
  atomicAdd(&sum[(size_t)g * 256 + c], v);
  atomicMaxF(&pmax[(size_t)g * 256 + c], v);
  if (c == 0) atomicAdd(&cnt[g], 1.f);
}

__global__ __launch_bounds__(THREADS) void k_pool_fin(
    const float* __restrict__ sum, const float* __restrict__ pmax,
    const float* __restrict__ cnt, float* __restrict__ hg, _Float16* __restrict__ hgf, int G) {
  int i = blockIdx.x * THREADS + threadIdx.x;
  if (i >= G * 256) return;
  int g = i >> 8, c = i & 255;
  float ct = cnt[g];
  float mean = sum[i] / fmaxf(ct, 1.f);
  float mx = (ct > 0.f) ? pmax[i] : 0.f;
  size_t o0 = (size_t)g * 512 + c;
  size_t o1 = o0 + 256;
  hg[o0] = mean; hg[o1] = mx;
  hgf[o0] = (_Float16)mean; hgf[o1] = (_Float16)mx;
}

// ---------------------------------------------------------------------------
extern "C" void kernel_launch(void* const* d_in, const int* in_sizes, int n_in,
                              void* d_out, int out_size, void* d_ws, size_t ws_size,
                              hipStream_t stream) {
  const int N = in_sizes[0] / 64;
  const int E = in_sizes[1] / 2;
  const int G = out_size / 12;
  const int Etot = E + N;

  const float* x      = (const float*)d_in[0];
  const int*   ei     = (const int*)d_in[1];
  const int*   batch  = (const int*)d_in[2];
  const float* W_in   = (const float*)d_in[3];
  const float* b_in   = (const float*)d_in[4];
  const float* gW[3]  = {(const float*)d_in[5],  (const float*)d_in[11], (const float*)d_in[17]};
  const float* gas[3] = {(const float*)d_in[6],  (const float*)d_in[12], (const float*)d_in[18]};
  const float* gad[3] = {(const float*)d_in[7],  (const float*)d_in[13], (const float*)d_in[19]};
  const float* bng[3] = {(const float*)d_in[9],  (const float*)d_in[15], (const float*)d_in[21]};
  const float* bnb[3] = {(const float*)d_in[10], (const float*)d_in[16], (const float*)d_in[22]};
  const float* mW1 = (const float*)d_in[23];
  const float* mg1 = (const float*)d_in[25];
  const float* mbe1 = (const float*)d_in[26];
  const float* mW2 = (const float*)d_in[27];
  const float* mg2 = (const float*)d_in[29];
  const float* mbe2 = (const float*)d_in[30];
  const float* hW = (const float*)d_in[31];
  const float* hb = (const float*)d_in[32];

  // workspace arena (256B aligned carves)
  char* ws = (char*)d_ws;
  size_t off = 0;
  auto carve = [&](size_t bytes) -> char* {
    char* p = ws + off;
    off += (bytes + 255) & ~(size_t)255;
    return p;
  };

  _Float16* x_f16   = (_Float16*)carve((size_t)N * 64 * 2);
  float*    hshort  = (float*)carve((size_t)N * 256 * 4);
  float*    h       = (float*)carve((size_t)N * 256 * 4);
  _Float16* h_f16   = (_Float16*)carve((size_t)N * 256 * 2);
  float*    xw      = (float*)carve((size_t)N * 256 * 4);
  float*    agg     = (float*)carve((size_t)N * 256 * 4);
  float*    es      = (float*)carve((size_t)N * 4 * 4);
  float*    ed      = (float*)carve((size_t)N * 4 * 4);
  float*    mbuf    = (float*)carve((size_t)N * 4 * 4);
  float*    den     = (float*)carve((size_t)N * 4 * 4);
  float*    exbuf   = (float*)carve((size_t)Etot * 4 * 4);
  _Float16* WtIn    = (_Float16*)carve(256 * 64 * 2);
  _Float16* gWt[3];
  gWt[0] = (_Float16*)carve(256 * 64 * 2);
  gWt[1] = (_Float16*)carve(256 * 256 * 2);
  gWt[2] = (_Float16*)carve(256 * 256 * 2);
  _Float16* m1t     = (_Float16*)carve(512 * 512 * 2);
  _Float16* m2t     = (_Float16*)carve(256 * 512 * 2);
  _Float16* ht      = (_Float16*)carve(16 * 256 * 2);
  float*    mu      = (float*)carve(512 * 4);
  float*    rs      = (float*)carve(512 * 4);
  float*    psum    = (float*)carve((size_t)G * 256 * 4);
  float*    pmax    = (float*)carve((size_t)G * 256 * 4);
  float*    pcnt    = (float*)carve((size_t)G * 4);
  float*    hg      = (float*)carve((size_t)G * 512 * 4);
  _Float16* hg_f16  = (_Float16*)carve((size_t)G * 512 * 2);
  float*    s1      = (float*)carve((size_t)G * 512 * 4);
  _Float16* s1_f16  = (_Float16*)carve((size_t)G * 512 * 2);
  float*    s2      = (float*)carve((size_t)G * 256 * 4);
  _Float16* s2_f16  = (_Float16*)carve((size_t)G * 256 * 2);
  (void)ws_size; (void)n_in;

  auto blocks = [](size_t n) { return (unsigned)((n + THREADS - 1) / THREADS); };
  auto gemm = [&](const _Float16* A, const _Float16* Bt, const float* bias, float* Out,
                  int M, int K, int Npad, int Nreal, int ldo) {
    int Mt = (M + 15) / 16, Nt = Npad / 16;
    int Ntg = (Nt + 3) / 4;
    int nb = (Mt * Ntg + 7) / 8;
    k_wmma_gemm<<<nb, THREADS, 0, stream>>>(A, Bt, bias, Out, M, K, Npad, Nreal, ldo);
  };

  // ---- prep: conversions + weight transposes
  k_cvt_f16<<<blocks((size_t)N * 64), THREADS, 0, stream>>>(x, x_f16, (size_t)N * 64);
  k_transpose_w<<<blocks(256 * 64), THREADS, 0, stream>>>(W_in,  WtIn,   64, 256, 256);
  k_transpose_w<<<blocks(256 * 64), THREADS, 0, stream>>>(gW[0], gWt[0], 64, 256, 256);
  k_transpose_w<<<blocks(256 * 256), THREADS, 0, stream>>>(gW[1], gWt[1], 256, 256, 256);
  k_transpose_w<<<blocks(256 * 256), THREADS, 0, stream>>>(gW[2], gWt[2], 256, 256, 256);
  k_transpose_w<<<blocks(512 * 512), THREADS, 0, stream>>>(mW1, m1t, 512, 512, 512);
  k_transpose_w<<<blocks(256 * 512), THREADS, 0, stream>>>(mW2, m2t, 512, 256, 256);
  k_transpose_w<<<blocks(16 * 256), THREADS, 0, stream>>>(hW, ht, 256, 12, 16);

  // ---- input shortcut: h_short = x @ W_in + b_in
  gemm(x_f16, WtIn, b_in, hshort, N, 64, 256, 256, 256);

  // ---- 3 GAT layers
  for (int l = 0; l < 3; ++l) {
    const _Float16* A = (l == 0) ? x_f16 : h_f16;
    const int K = (l == 0) ? 64 : 256;
    gemm(A, gWt[l], nullptr, xw, N, K, 256, 256, 256);
    k_head_dots<<<blocks((size_t)N * 4), THREADS, 0, stream>>>(xw, gas[l], gad[l], es, ed, N);

    k_fill<<<blocks((size_t)N * 4), THREADS, 0, stream>>>(mbuf, -1e30f, (size_t)N * 4);
    k_fill<<<blocks((size_t)N * 4), THREADS, 0, stream>>>(den, 0.f, (size_t)N * 4);
    k_fill<<<blocks((size_t)N * 256), THREADS, 0, stream>>>(agg, 0.f, (size_t)N * 256);

    k_edge_max<<<blocks((size_t)Etot * 4), THREADS, 0, stream>>>(ei, es, ed, mbuf, E, Etot);
    k_edge_exp<<<blocks((size_t)Etot * 4), THREADS, 0, stream>>>(ei, es, ed, mbuf, exbuf, den, E, Etot);
    k_edge_agg<<<Etot, THREADS, 0, stream>>>(ei, exbuf, den, xw, agg, E, Etot);

    k_col_stats<<<256, THREADS, 0, stream>>>(agg, N, 256, mu, rs);
    const float* res = (l == 0) ? hshort : h;
    k_gat_post<<<blocks((size_t)N * 256), THREADS, 0, stream>>>(
        agg, res, mu, rs, bng[l], bnb[l], h, h_f16, (size_t)N * 256, 255);
  }

  // ---- mean/max pooling per graph
  k_fill<<<blocks((size_t)G * 256), THREADS, 0, stream>>>(psum, 0.f, (size_t)G * 256);
  k_fill<<<blocks((size_t)G * 256), THREADS, 0, stream>>>(pmax, -1e30f, (size_t)G * 256);
  k_fill<<<blocks((size_t)G), THREADS, 0, stream>>>(pcnt, 0.f, (size_t)G);
  k_pool_acc<<<blocks((size_t)N * 256), THREADS, 0, stream>>>(h, batch, psum, pmax, pcnt, N);
  k_pool_fin<<<blocks((size_t)G * 256), THREADS, 0, stream>>>(psum, pmax, pcnt, hg, hg_f16, G);

  // ---- MLP head (biases before BN cancel; only hb matters)
  gemm(hg_f16, m1t, nullptr, s1, G, 512, 512, 512, 512);
  k_col_stats<<<512, THREADS, 0, stream>>>(s1, G, 512, mu, rs);
  k_relu_post<<<blocks((size_t)G * 512), THREADS, 0, stream>>>(
      s1, mu, rs, mg1, mbe1, s1_f16, (size_t)G * 512, 511);

  gemm(s1_f16, m2t, nullptr, s2, G, 512, 256, 256, 256);
  k_col_stats<<<256, THREADS, 0, stream>>>(s2, G, 256, mu, rs);
  k_relu_post<<<blocks((size_t)G * 256), THREADS, 0, stream>>>(
      s2, mu, rs, mg2, mbe2, s2_f16, (size_t)G * 256, 255);

  gemm(s2_f16, ht, hb, (float*)d_out, G, 256, 16, 12, 12);
}
